// AttnEncoder_4999341932804
// MI455X (gfx1250) — matrix operand order repaired and checked by reading, not compiled, not run on hardware
//
#include <hip/hip_runtime.h>
#include <hip/hip_bf16.h>
#include <stdint.h>

// Problem constants (from reference): B=512, T=512, N=64, H=128
#define Bsz   512
#define Tsz   512
#define Nsz   64
#define Hsz   128
#define Kdim  192     // N + H : GEMM K dimension (concat [weighted | h])
#define ZN    512     // 4H    : GEMM N dimension (i,f,g,o gates)
#define MB    16      // batch rows per block = one WMMA M tile
#define NWAVE 8       // waves per block (wave32 -> 256 threads)
#define NT    4       // 16-wide N tiles per wave (8 waves * 4 * 16 = 512)
#define KT    6       // 32-deep K tiles (6 * 32 = 192)
#define ASTRIDE 200   // padded A-staging row stride in bf16 elems (mult of 8)

typedef __attribute__((ext_vector_type(16))) __bf16        v16bf;
typedef __attribute__((ext_vector_type(8)))  float         v8f;
typedef __attribute__((ext_vector_type(8)))  unsigned short v8us;

union BF16x16 {
    unsigned short u[16];
    v16bf v;
    struct { v8us lo, hi; } h;
};

// round-to-nearest-even f32 -> bf16 (bit pattern)
__device__ __forceinline__ unsigned short f2bf(float f) {
    uint32_t u = __float_as_uint(f);
    uint32_t r = u + 0x7FFFu + ((u >> 16) & 1u);
    return (unsigned short)(r >> 16);
}
__device__ __forceinline__ float sigm(float x) { return 1.0f / (1.0f + __expf(-x)); }

__global__ __launch_bounds__(256)
void attn_lstm_persistent(const float* __restrict__ input,   // (B,T,N)
                          const float* __restrict__ h0,      // (B,H)
                          const float* __restrict__ c0,      // (B,H)
                          const float* __restrict__ W_ih,    // (4H,N)
                          const float* __restrict__ W_hh,    // (4H,H)
                          const float* __restrict__ b_ih,    // (4H)
                          const float* __restrict__ b_hh,    // (4H)
                          const float* __restrict__ attn_w,  // (2H+T) = 768
                          const float* __restrict__ attn_b,  // (1)
                          float* __restrict__ out_attn,      // (B,T,N)
                          float* __restrict__ out_enc)       // (B,T,H)
{
    // ---- LDS (62.7 KB total: fits any LDS-limit interpretation) ----
    __shared__ float h_s[MB * Hsz];          //  8 KB carried hidden state
    __shared__ float c_s[MB * Hsz];          //  8 KB carried cell state
    __shared__ float z_s[MB * ZN];           // 32 KB gate pre-activations (also w_x staging in init)
    __shared__ float xc_s[MB * Nsz];         //  4 KB x_contrib
    __shared__ float whc_s[2 * Hsz];         //  1 KB w_h | w_c
    __shared__ float bias_s[ZN];             //  2 KB b_ih + b_hh
    __shared__ __align__(16) unsigned short a_s[MB * ASTRIDE]; // 6.4 KB bf16 A staging

    const int tid   = threadIdx.x;
    const int wave  = tid >> 5;
    const int lane  = tid & 31;
    const int brow0 = blockIdx.x * MB;

    // ================= one-time init =================
    if (tid < 256) {
        whc_s[tid]       = attn_w[tid];                    // w_h (0..127) | w_c (128..255)
        bias_s[tid]      = b_ih[tid]       + b_hh[tid];
        bias_s[tid+256]  = b_ih[tid+256]   + b_hh[tid+256];
        z_s[tid]         = attn_w[256 + tid];              // stage w_x[0..255]
        z_s[tid+256]     = attn_w[512 + tid];              // stage w_x[256..511]
    }
    for (int idx = tid; idx < MB * Hsz; idx += 256) {
        int r = idx >> 7, j = idx & 127;
        h_s[idx] = h0[(size_t)(brow0 + r) * Hsz + j];
        c_s[idx] = c0[(size_t)(brow0 + r) * Hsz + j];
    }
    __syncthreads();

    // x_contrib[b][n] = sum_t input[b,t,n] * w_x[t] + attn_b
    {
        const float ab = attn_b[0];
        const int n  = tid & 63;
        const int r0 = tid >> 6;                 // 0..3 ; rows r0, r0+4, r0+8, r0+12
        float acc[4] = {ab, ab, ab, ab};
        for (int t = 0; t < Tsz; ++t) {
            const float wx = z_s[t];
            #pragma unroll
            for (int q = 0; q < 4; ++q)
                acc[q] += input[(((size_t)(brow0 + r0 + q*4)) * Tsz + t) * Nsz + n] * wx;
        }
        #pragma unroll
        for (int q = 0; q < 4; ++q) xc_s[(r0 + q*4) * Nsz + n] = acc[q];
    }

    // Load weight B-fragments into VGPRs once (bf16), per documented B layout:
    // lane L holds column n = L%16; elements 0-7: K 8h..8h+7, 8-15: K 16+8h..16+8h+7 (h = L/16).
    // Wcat^T[k][n] = k < 64 ? W_ih[n][k] : W_hh[n][k-64]
    v16bf Bf[KT][NT];
    #pragma unroll
    for (int kt = 0; kt < KT; ++kt) {
        const int kb = kt * 32 + (lane >> 4) * 8;
        #pragma unroll
        for (int j = 0; j < NT; ++j) {
            const int nbase = (wave * NT + j) * 16;
            BF16x16 u;
            #pragma unroll
            for (int i = 0; i < 16; ++i) {
                const int n = nbase + (lane & 15);
                const int k = kb + (i < 8 ? i : 8 + i);
                const float w = (k < Nsz) ? W_ih[(size_t)n * Nsz + k]
                                          : W_hh[(size_t)n * Hsz + (k - Nsz)];
                u.u[i] = f2bf(w);
            }
            Bf[kt][j] = u.v;
        }
    }
    __syncthreads();  // z_s (w_x) reads + xc_s/a-state writes complete

    // ================= sequential time loop =================
    for (int t = 0; t < Tsz; ++t) {
        // ---- Phase A+B: attention scalar, softmax, weighted, A-staging ----
        {
            const int r   = 2 * wave + (lane >> 4);  // each 16-lane half owns a row
            const int sub = lane & 15;

            // s = h . w_h + c . w_c  (16 lanes x 8 MACs each, butterfly reduce)
            float s = 0.f;
            const int k0 = sub * 8;
            #pragma unroll
            for (int q = 0; q < 8; ++q) {
                s += h_s[r * Hsz + k0 + q] * whc_s[k0 + q];
                s += c_s[r * Hsz + k0 + q] * whc_s[Hsz + k0 + q];
            }
            s += __shfl_xor(s, 1, 32);
            s += __shfl_xor(s, 2, 32);
            s += __shfl_xor(s, 4, 32);
            s += __shfl_xor(s, 8, 32);

            // softmax over 64 cols (4 per lane)
            const int cb = sub * 4;
            float e0 = s + xc_s[r * Nsz + cb + 0];
            float e1 = s + xc_s[r * Nsz + cb + 1];
            float e2 = s + xc_s[r * Nsz + cb + 2];
            float e3 = s + xc_s[r * Nsz + cb + 3];
            float mx = fmaxf(fmaxf(e0, e1), fmaxf(e2, e3));
            mx = fmaxf(mx, __shfl_xor(mx, 1, 32));
            mx = fmaxf(mx, __shfl_xor(mx, 2, 32));
            mx = fmaxf(mx, __shfl_xor(mx, 4, 32));
            mx = fmaxf(mx, __shfl_xor(mx, 8, 32));
            float p0 = __expf(e0 - mx), p1 = __expf(e1 - mx);
            float p2 = __expf(e2 - mx), p3 = __expf(e3 - mx);
            float sm = p0 + p1 + p2 + p3;
            sm += __shfl_xor(sm, 1, 32);
            sm += __shfl_xor(sm, 2, 32);
            sm += __shfl_xor(sm, 4, 32);
            sm += __shfl_xor(sm, 8, 32);
            const float inv = 1.0f / sm;
            const float a0 = p0 * inv, a1 = p1 * inv, a2 = p2 * inv, a3 = p3 * inv;

            const size_t base = ((size_t)(brow0 + r) * Tsz + t) * Nsz + cb;
            const float4 xv = *(const float4*)(input + base);
            *(float4*)(out_attn + base) = make_float4(a0, a1, a2, a3);
            if (t + 1 < Tsz) __builtin_prefetch(input + base + Nsz, 0, 1);

            // stage weighted (cols 0..63) and h (cols 64..191) as bf16 A matrix
            a_s[r * ASTRIDE + cb + 0] = f2bf(a0 * xv.x);
            a_s[r * ASTRIDE + cb + 1] = f2bf(a1 * xv.y);
            a_s[r * ASTRIDE + cb + 2] = f2bf(a2 * xv.z);
            a_s[r * ASTRIDE + cb + 3] = f2bf(a3 * xv.w);
            #pragma unroll
            for (int q = 0; q < 8; ++q)
                a_s[r * ASTRIDE + Nsz + k0 + q] = f2bf(h_s[r * Hsz + k0 + q]);
        }
        __syncthreads();

        // ---- Phase C: GEMM z = A(16x192) * Wcat^T(192x512) + bias ----
        {
            // A-fragment layout (16-bit, 16x32): lane m (0..15): K 0-7,16-23 ;
            // lane m+16: K 8-15,24-31  -> two 16B LDS vector loads each.
            v16bf Af[KT];
            const int m_   = lane & 15;
            const int half = lane >> 4;
            #pragma unroll
            for (int kt = 0; kt < KT; ++kt) {
                const int kk = kt * 32 + half * 8;
                BF16x16 u;
                u.h.lo = *(const v8us*)&a_s[m_ * ASTRIDE + kk];
                u.h.hi = *(const v8us*)&a_s[m_ * ASTRIDE + kk + 16];
                Af[kt] = u.v;
            }
            const int nl    = lane & 15;
            const int rbase = (lane >> 4) * 8;  // C/D: lanes 0-15 -> M 0-7, 16-31 -> M 8-15
            #pragma unroll
            for (int j = 0; j < NT; ++j) {
                const int ncol = (wave * NT + j) * 16 + nl;
                const float bv = bias_s[ncol];
                v8f acc = {bv, bv, bv, bv, bv, bv, bv, bv};
                #pragma unroll
                for (int kt = 0; kt < KT; ++kt)
                    acc = __builtin_amdgcn_wmma_f32_16x16x32_bf16(
                        false, Af[kt], false, Bf[kt][j], (short)0, acc, false, false);
                #pragma unroll
                for (int v = 0; v < 8; ++v)
                    z_s[(rbase + v) * ZN + ncol] = acc[v];
            }
        }
        __syncthreads();

        // ---- Phase D: LSTM cell update (i,f,g,o), emit h_new ----
        #pragma unroll
        for (int q = 0; q < (MB * Hsz) / 256; ++q) {   // 8 elements per thread
            const int idx = q * 256 + tid;
            const int rr = idx >> 7, jj = idx & 127;
            const float zi = z_s[rr * ZN + jj];
            const float zf = z_s[rr * ZN + Hsz     + jj];
            const float zg = z_s[rr * ZN + 2 * Hsz + jj];
            const float zo = z_s[rr * ZN + 3 * Hsz + jj];
            const float cp = c_s[idx];
            const float cn = sigm(zf) * cp + sigm(zi) * tanhf(zg);
            const float hn = sigm(zo) * tanhf(cn);
            c_s[idx] = cn;
            h_s[idx] = hn;
            out_enc[((size_t)(brow0 + rr) * Tsz + t) * Hsz + jj] = hn;
        }
        __syncthreads();
    }
}

extern "C" void kernel_launch(void* const* d_in, const int* in_sizes, int n_in,
                              void* d_out, int out_size, void* d_ws, size_t ws_size,
                              hipStream_t stream) {
    (void)in_sizes; (void)n_in; (void)out_size; (void)d_ws; (void)ws_size;
    const float* input  = (const float*)d_in[0];
    const float* h0     = (const float*)d_in[1];
    const float* c0     = (const float*)d_in[2];
    const float* W_ih   = (const float*)d_in[3];
    const float* W_hh   = (const float*)d_in[4];
    const float* b_ih   = (const float*)d_in[5];
    const float* b_hh   = (const float*)d_in[6];
    const float* attn_w = (const float*)d_in[7];
    const float* attn_b = (const float*)d_in[8];

    float* out_attn = (float*)d_out;                                   // (B,T,N)
    float* out_enc  = out_attn + (size_t)Bsz * Tsz * Nsz;              // (B,T,H)

    dim3 grid(Bsz / MB);   // 32 persistent blocks, one 16-row batch slice each
    dim3 block(256);       // 8 wave32 waves
    hipLaunchKernelGGL(attn_lstm_persistent, grid, block, 0, stream,
                       input, h0, c0, W_ih, W_hh, b_ih, b_hh,
                       attn_w, attn_b, out_attn, out_enc);
}
